// LSTM_48180943127338
// MI455X (gfx1250) — matrix operand order repaired and checked by reading, not compiled, not run on hardware
//
#include <hip/hip_runtime.h>

#define SEQ   8192
#define ISZ   512
#define HSZ   2048
#define G4H   (4 * HSZ)   // 8192
#define NBLK  512         // persistent blocks, 1 wave each

typedef __attribute__((ext_vector_type(16))) __bf16 v16bf;
typedef __attribute__((ext_vector_type(8)))  __bf16 v8bf;
typedef __attribute__((ext_vector_type(8)))  float  v8f;

__device__ __forceinline__ unsigned short f32_to_bf16_rne(float f) {
    unsigned u = __float_as_uint(f);
    u += 0x7FFFu + ((u >> 16) & 1u);
    return (unsigned short)(u >> 16);
}

__device__ __forceinline__ float sigm(float x) { return 1.0f / (1.0f + expf(-x)); }

// ---------------------------------------------------------------- conversions
__global__ void cvt_f32_bf16(const float* __restrict__ src,
                             unsigned short* __restrict__ dst, int n) {
    int i = blockIdx.x * blockDim.x + threadIdx.x;
    if (i < n) dst[i] = f32_to_bf16_rne(src[i]);
}

// ------------------------------------------------------- x_proj GEMM (WMMA)
// xp[m, j] = sum_k X[m,k] * W[j,k] + b_ih[j] + b_hh[j]
// One wave per 16x16 output tile. A = X rows (16x32 bf16), B = W rows as
// columns of a 32x16 tile (per-lane: two runs of 8 K values, 16 apart).
__global__ __launch_bounds__(256) void xproj_gemm(
    const unsigned short* __restrict__ Xb,   // [SEQ, ISZ] bf16
    const unsigned short* __restrict__ Wb,   // [G4H, ISZ] bf16
    const float* __restrict__ b_ih, const float* __restrict__ b_hh,
    float* __restrict__ xp)                  // [SEQ, G4H] f32
{
    const int wave = blockIdx.x * (blockDim.x >> 5) + (threadIdx.x >> 5);
    const int lane = threadIdx.x & 31;
    const int lrow = lane & 15;
    const int half = lane >> 4;
    const int ntn  = G4H / 16;               // 512
    const int tm   = wave / ntn;
    const int tn   = wave % ntn;
    const int m0   = tm * 16, j0 = tn * 16;

    const __bf16* arow = (const __bf16*)Xb + (size_t)(m0 + lrow) * ISZ;
    const __bf16* brow = (const __bf16*)Wb + (size_t)(j0 + lrow) * ISZ;

    v8f acc;
#pragma unroll
    for (int i = 0; i < 8; ++i) acc[i] = 0.0f;

#pragma unroll 4
    for (int k = 0; k < ISZ; k += 32) {
        // A 16x32 bf16: lanes 0-15 hold K {0..7, 16..23}; lanes 16-31 hold {8..15, 24..31}
        v8bf alo = *(const v8bf*)(arow + k + half * 8);
        v8bf ahi = *(const v8bf*)(arow + k + 16 + half * 8);
        // B 32x16 bf16: lanes 0-15 hold K {0..7, 16..23} of column lrow; lanes 16-31 hold {8..15, 24..31}
        v8bf blo = *(const v8bf*)(brow + k + half * 8);
        v8bf bhi = *(const v8bf*)(brow + k + 16 + half * 8);
        v16bf a, b;
#pragma unroll
        for (int i = 0; i < 8; ++i) {
            a[i] = alo[i]; a[i + 8] = ahi[i];
            b[i] = blo[i]; b[i + 8] = bhi[i];
        }
        acc = __builtin_amdgcn_wmma_f32_16x16x32_bf16(false, a, false, b,
                                                      (short)0, acc, false, false);
    }

    const int   j    = j0 + lrow;
    const float bias = b_ih[j] + b_hh[j];
#pragma unroll
    for (int r = 0; r < 8; ++r) {
        const int m = m0 + half * 8 + r;     // C/D layout: lane<16 -> M=r, lane>=16 -> M=8+r
        xp[(size_t)m * G4H + j] = acc[r] + bias;
    }
}

// ---------------------------------------------------------- grid spin barrier
__device__ __forceinline__ void grid_barrier(unsigned* counter, unsigned barno) {
    __threadfence();
    if (threadIdx.x == 0) {
        atomicAdd(counter, 1u);
        const unsigned target = barno * (unsigned)NBLK;
        while (__hip_atomic_load(counter, __ATOMIC_RELAXED, __HIP_MEMORY_SCOPE_AGENT) < target) {
            __builtin_amdgcn_s_sleep(1);
        }
    }
    __threadfence();
}

// --------------------------------------------------- persistent LSTM scan
// 512 blocks x 1 wave. Wave w owns gate rows [jb, jb+16) with
// q = w>>7 (gate quadrant i/f/g/o), tile = w&127. Phase 1: WMMA GEMV
// (B has h broadcast into column 0 only -> D column 0 = dot products).
// Phase 2: block b updates hidden elements [4b, 4b+4).
__global__ __launch_bounds__(32) void lstm_recurrent(
    const unsigned short* __restrict__ Whh,  // [G4H, HSZ] bf16
    const unsigned short* __restrict__ Wih,  // [G4H, ISZ] bf16 (fused path)
    const unsigned short* __restrict__ Xb,   // [SEQ, ISZ] bf16 (fused path)
    const float* __restrict__ xp,            // [SEQ, G4H] f32 (precomputed path)
    const float* __restrict__ b_ih, const float* __restrict__ b_hh,
    float* __restrict__ gates,               // [G4H] raw recurrent dots
    float* __restrict__ cbuf,                // [HSZ]
    float* __restrict__ hbuf,                // [HSZ] f32 (for final fc)
    unsigned short* __restrict__ hb,         // [HSZ] bf16 (WMMA B operand)
    unsigned* __restrict__ counter, int use_xp)
{
    const int w    = blockIdx.x;
    const int lane = threadIdx.x;
    const int lrow = lane & 15;
    const int half = lane >> 4;
    const int q    = w >> 7;
    const int tile = w & 127;
    const int jb   = q * HSZ + tile * 16;

    const __bf16* Arow = (const __bf16*)Whh + (size_t)(jb + lrow) * HSZ;
    const __bf16* Irow = (const __bf16*)Wih + (size_t)(jb + lrow) * ISZ;
    const __bf16* hbf  = (const __bf16*)hb;
    const bool    col0 = (lrow == 0);
    const __bf16  zero = (__bf16)0.0f;
    const int     jj   = blockIdx.x * 4;     // phase-2 ownership

    unsigned bar = 0;

    for (int t = 0; t < SEQ; ++t) {
        v8f acc;
#pragma unroll
        for (int i = 0; i < 8; ++i) acc[i] = 0.0f;

        if (!use_xp) {  // fused input projection: += W_ih[jb..] . x_t
            const __bf16* xrow = (const __bf16*)Xb + (size_t)t * ISZ;
#pragma unroll 2
            for (int k = 0; k < ISZ; k += 32) {
                v8bf alo = *(const v8bf*)(Irow + k + half * 8);
                v8bf ahi = *(const v8bf*)(Irow + k + 16 + half * 8);
                v8bf x0  = *(const v8bf*)(xrow + k + half * 8);
                v8bf x1  = *(const v8bf*)(xrow + k + 16 + half * 8);
                v16bf a, b;
#pragma unroll
                for (int i = 0; i < 8; ++i) {
                    a[i] = alo[i]; a[i + 8] = ahi[i];
                    b[i]     = col0 ? x0[i] : zero;
                    b[i + 8] = col0 ? x1[i] : zero;
                }
                acc = __builtin_amdgcn_wmma_f32_16x16x32_bf16(false, a, false, b,
                                                              (short)0, acc, false, false);
            }
        }

#pragma unroll 2
        for (int k = 0; k < HSZ; k += 32) {  // += W_hh[jb..] . h
            v8bf alo = *(const v8bf*)(Arow + k + half * 8);
            v8bf ahi = *(const v8bf*)(Arow + k + 16 + half * 8);
            v8bf h0  = *(const v8bf*)(hbf + k + half * 8);
            v8bf h1  = *(const v8bf*)(hbf + k + 16 + half * 8);
            v16bf a, b;
#pragma unroll
            for (int i = 0; i < 8; ++i) {
                a[i] = alo[i]; a[i + 8] = ahi[i];
                b[i]     = col0 ? h0[i] : zero;
                b[i + 8] = col0 ? h1[i] : zero;
            }
            acc = __builtin_amdgcn_wmma_f32_16x16x32_bf16(false, a, false, b,
                                                          (short)0, acc, false, false);
        }

        if (col0) {  // D column 0: lane 0 -> rows 0-7, lane 16 -> rows 8-15
#pragma unroll
            for (int r = 0; r < 8; ++r) gates[jb + half * 8 + r] = acc[r];
        }

        grid_barrier(counter, ++bar);

        if (lane < 4) {  // elementwise cell update for 4 hidden units
            const int j = jj + lane;
            float gi = gates[j];
            float gf = gates[j + HSZ];
            float gg = gates[j + 2 * HSZ];
            float go = gates[j + 3 * HSZ];
            if (use_xp) {
                const float* xpt = xp + (size_t)t * G4H;
                gi += xpt[j]; gf += xpt[j + HSZ];
                gg += xpt[j + 2 * HSZ]; go += xpt[j + 3 * HSZ];
            } else {
                gi += b_ih[j]           + b_hh[j];
                gf += b_ih[j + HSZ]     + b_hh[j + HSZ];
                gg += b_ih[j + 2 * HSZ] + b_hh[j + 2 * HSZ];
                go += b_ih[j + 3 * HSZ] + b_hh[j + 3 * HSZ];
            }
            float c = sigm(gf) * cbuf[j] + sigm(gi) * tanhf(gg);
            cbuf[j] = c;
            float h = sigm(go) * tanhf(c);
            hbuf[j] = h;
            hb[j]   = f32_to_bf16_rne(h);
        }

        grid_barrier(counter, ++bar);
    }
}

// ------------------------------------------------------------------ final fc
__global__ void fc_kernel(const float* __restrict__ h, const float* __restrict__ fcw,
                          const float* __restrict__ fcb, float* __restrict__ out) {
    __shared__ float red[256];
    float s = 0.0f;
    for (int j = threadIdx.x; j < HSZ; j += 256) s += h[j] * fcw[j];
    red[threadIdx.x] = s;
    __syncthreads();
    for (int off = 128; off > 0; off >>= 1) {
        if ((int)threadIdx.x < off) red[threadIdx.x] += red[threadIdx.x + off];
        __syncthreads();
    }
    if (threadIdx.x == 0) out[0] = red[0] + fcb[0];
}

// -------------------------------------------------------------------- launch
extern "C" void kernel_launch(void* const* d_in, const int* in_sizes, int n_in,
                              void* d_out, int out_size, void* d_ws, size_t ws_size,
                              hipStream_t stream) {
    (void)in_sizes; (void)n_in; (void)out_size;
    const float* input = (const float*)d_in[0];
    const float* W_ih  = (const float*)d_in[1];
    const float* W_hh  = (const float*)d_in[2];
    const float* b_ih  = (const float*)d_in[3];
    const float* b_hh  = (const float*)d_in[4];
    const float* fc_w  = (const float*)d_in[5];
    const float* fc_b  = (const float*)d_in[6];

    char*  ws  = (char*)d_ws;
    size_t off = 0;
    auto carve = [&](size_t bytes) {
        char* p = ws + off;
        off = (off + bytes + 255) & ~(size_t)255;
        return p;
    };
    unsigned short* Wih_b = (unsigned short*)carve((size_t)G4H * ISZ * 2);
    unsigned short* Xb    = (unsigned short*)carve((size_t)SEQ * ISZ * 2);
    unsigned short* Whh_b = (unsigned short*)carve((size_t)G4H * HSZ * 2);
    float*          gates = (float*)carve((size_t)G4H * 4);
    float*          cbuf  = (float*)carve((size_t)HSZ * 4);
    float*          hbuf  = (float*)carve((size_t)HSZ * 4);
    unsigned short* hb    = (unsigned short*)carve((size_t)HSZ * 2);
    unsigned*       counter = (unsigned*)carve(256);
    float*          xp    = (float*)(ws + off);
    const size_t    xp_bytes = (size_t)SEQ * G4H * 4;   // 256 MB
    const int       use_xp   = (off + xp_bytes <= ws_size) ? 1 : 0;

    hipMemsetAsync(counter, 0, 4, stream);
    hipMemsetAsync(cbuf, 0, (size_t)HSZ * 4, stream);
    hipMemsetAsync(hb,   0, (size_t)HSZ * 2, stream);

    const int n1 = G4H * ISZ;
    cvt_f32_bf16<<<(n1 + 255) / 256, 256, 0, stream>>>(W_ih, Wih_b, n1);
    const int n2 = SEQ * ISZ;
    cvt_f32_bf16<<<(n2 + 255) / 256, 256, 0, stream>>>(input, Xb, n2);
    const int n3 = G4H * HSZ;
    cvt_f32_bf16<<<(n3 + 255) / 256, 256, 0, stream>>>(W_hh, Whh_b, n3);

    if (use_xp) {
        const int nwaves = (SEQ / 16) * (G4H / 16);     // 262144 tiles
        xproj_gemm<<<nwaves / 8, 256, 0, stream>>>(Xb, Wih_b, b_ih, b_hh, xp);
    }

    lstm_recurrent<<<NBLK, 32, 0, stream>>>(Whh_b, Wih_b, Xb, xp, b_ih, b_hh,
                                            gates, cbuf, hbuf, hb, counter, use_xp);

    fc_kernel<<<1, 256, 0, stream>>>(hbuf, fc_w, fc_b, (float*)d_out);
}
